// PoseEncoder_85134841741850
// MI455X (gfx1250) — matrix operand order repaired and checked by reading, not compiled, not run on hardware
//
#include <hip/hip_runtime.h>

typedef __attribute__((ext_vector_type(16))) _Float16 v16h;
typedef __attribute__((ext_vector_type(8)))  _Float16 v8h;
typedef __attribute__((ext_vector_type(8)))  float    v8f;

#define NKPT  17
#define INDIM 34      // 17 * 2
#define KPAD1 64      // K padded for GEMM1
#define HID   128
#define ODIM  64
#define WAVES 8
#define RPW   32                // rows per wave
#define RPB   (WAVES * RPW)     // 256 rows per block
#define XS    72                // LDS stride (halfs) for staged X rows  (144B, 16B aligned)
#define HS    136               // LDS stride (halfs) for H rows         (272B, 16B aligned)

struct SMem {
  _Float16 w1[HID][XS];        // W1 transposed: [n][k], k<34 real, rest 0
  _Float16 w2[ODIM][HS];       // W2 transposed: [n][k], k<128
  float    bb1[HID];
  float    bb2[ODIM];
  _Float16 x[WAVES][RPW][XS];  // centered inputs, f16, K padded to 64 with zeros
  _Float16 h[WAVES][RPW][HS];  // hidden activations, f16
};

// A-matrix fragment (16-bit, 16x32): lane holds K = off+[0..7] and off+16+[0..7]
__device__ __forceinline__ v16h frag_split(const _Float16* p, int kbase) {
  v8h lo = *(const v8h*)(p + kbase);
  v8h hi = *(const v8h*)(p + kbase + 16);
  return __builtin_shufflevector(lo, hi, 0,1,2,3,4,5,6,7,8,9,10,11,12,13,14,15);
}
// B-matrix fragment (16-bit, 32x16): lane holds K = off+[0..7] and off+16+[0..7]
__device__ __forceinline__ v16h frag_contig(const _Float16* p, int kbase) {
  v8h lo = *(const v8h*)(p + kbase);
  v8h hi = *(const v8h*)(p + kbase + 16);
  return __builtin_shufflevector(lo, hi, 0,1,2,3,4,5,6,7,8,9,10,11,12,13,14,15);
}

__global__ __launch_bounds__(RPB) void pose_encoder_wmma(
    const float* __restrict__ kp,  const float* __restrict__ W1,
    const float* __restrict__ pb1, const float* __restrict__ W2,
    const float* __restrict__ pb2, float* __restrict__ out, long long Ntot)
{
  __shared__ SMem sm;
  const int tid  = threadIdx.x;
  const int wave = tid >> 5;
  const int lane = tid & 31;
  const int n16  = lane & 15;
  const int hsel = lane >> 4;        // 0 for lanes 0-15, 1 for lanes 16-31
  const int offA = hsel * 8;         // A-frag K base offset
  const int offB = hsel * 8;         // B-frag K base offset
  const int moff = hsel * 8;         // C/D row offset

  // ---- stage weights (f32 -> f16, transposed, K-padded) ----
  for (int i = tid; i < HID * KPAD1; i += RPB) {
    int n = i >> 6, k = i & 63;
    sm.w1[n][k] = (k < INDIM) ? (_Float16)W1[k * HID + n] : (_Float16)0.0f;
  }
  for (int i = tid; i < ODIM * HID; i += RPB) {
    int n = i >> 7, k = i & 127;
    sm.w2[n][k] = (_Float16)W2[k * ODIM + n];
  }
  if (tid < HID)  sm.bb1[tid] = pb1[tid];
  if (tid < ODIM) sm.bb2[tid] = pb2[tid];
  __syncthreads();

  const long long rowbase = (long long)blockIdx.x * RPB + (long long)wave * RPW;

  // ---- per-lane: center one row, stage centered f16 into LDS ----
  {
    long long row = rowbase + lane;
    bool valid = row < Ntot;
    const float* p = kp + row * (NKPT * 3);
    float xs[NKPT], ys[NKPT];
#pragma unroll
    for (int i = 0; i < NKPT; ++i) {
      xs[i] = valid ? p[i * 3 + 0] : 0.0f;
      ys[i] = valid ? p[i * 3 + 1] : 0.0f;
    }
    float lx = xs[11], ly = ys[11], rx = xs[12], ry = ys[12];
    float hcx = 0.5f * (lx + rx), hcy = 0.5f * (ly + ry);
    bool hip_valid = ((fabsf(lx) + fabsf(ly)) > 0.0f) && ((fabsf(rx) + fabsf(ry)) > 0.0f);
    float cnt = 0.0f, sx = 0.0f, sy = 0.0f;
#pragma unroll
    for (int i = 0; i < NKPT; ++i) {
      float nz = ((fabsf(xs[i]) + fabsf(ys[i])) > 0.0f) ? 1.0f : 0.0f;
      cnt += nz; sx += xs[i] * nz; sy += ys[i] * nz;
    }
    float inv = 1.0f / fmaxf(cnt, 1.0f);
    float mmx = sx * inv, mmy = sy * inv;
    float fbx = (cnt > 0.0f) ? mmx : hcx;
    float fby = (cnt > 0.0f) ? mmy : hcy;
    float cx = hip_valid ? hcx : fbx;
    float cy = hip_valid ? hcy : fby;

    _Float16* dst = &sm.x[wave][lane][0];
#pragma unroll
    for (int i = 0; i < NKPT; ++i) {
      dst[2 * i]     = (_Float16)(xs[i] - cx);
      dst[2 * i + 1] = (_Float16)(ys[i] - cy);
    }
#pragma unroll
    for (int k = INDIM; k < KPAD1; ++k) dst[k] = (_Float16)0.0f;
  }
  asm volatile("s_wait_dscnt 0" ::: "memory");

  // ---- two 16-row tiles per wave ----
#pragma unroll
  for (int rt = 0; rt < 2; ++rt) {
    // GEMM1: X(16x64) @ W1(64x128) -> H(16x128), +b1, ReLU
    const _Float16* arow = &sm.x[wave][rt * 16 + n16][0];
    v16h a0 = frag_split(arow, offA);
    v16h a1 = frag_split(arow, offA + 32);
#pragma unroll
    for (int t = 0; t < 8; ++t) {
      const _Float16* bcol = &sm.w1[t * 16 + n16][0];
      v16h b0 = frag_contig(bcol, offB);
      v16h b1 = frag_contig(bcol, offB + 32);
      v8f acc = {};
      acc = __builtin_amdgcn_wmma_f32_16x16x32_f16(false, a0, false, b0, (short)0, acc, false, false);
      acc = __builtin_amdgcn_wmma_f32_16x16x32_f16(false, a1, false, b1, (short)0, acc, false, false);
      float bias = sm.bb1[t * 16 + n16];
#pragma unroll
      for (int r = 0; r < 8; ++r) {
        float v = acc[r] + bias;
        v = v > 0.0f ? v : 0.0f;
        sm.h[wave][rt * 16 + moff + r][t * 16 + n16] = (_Float16)v;
      }
    }
    asm volatile("s_wait_dscnt 0" ::: "memory");

    // GEMM2: H(16x128) @ W2(128x64) -> out(16x64), +b2
    const _Float16* hrow = &sm.h[wave][rt * 16 + n16][0];
#pragma unroll
    for (int t2 = 0; t2 < 4; ++t2) {
      v8f acc = {};
#pragma unroll
      for (int k = 0; k < 4; ++k) {
        v16h a = frag_split(hrow, offA + 32 * k);
        v16h b = frag_contig(&sm.w2[t2 * 16 + n16][0], offB + 32 * k);
        acc = __builtin_amdgcn_wmma_f32_16x16x32_f16(false, a, false, b, (short)0, acc, false, false);
      }
      float bias2 = sm.bb2[t2 * 16 + n16];
      long long wr = rowbase + rt * 16 + moff;
#pragma unroll
      for (int r = 0; r < 8; ++r) {
        long long row = wr + r;
        if (row < Ntot) out[row * ODIM + t2 * 16 + n16] = acc[r] + bias2;
      }
    }
  }
}

extern "C" void kernel_launch(void* const* d_in, const int* in_sizes, int n_in,
                              void* d_out, int out_size, void* d_ws, size_t ws_size,
                              hipStream_t stream) {
  const float* kp  = (const float*)d_in[0];
  const float* W1  = (const float*)d_in[1];
  const float* pb1 = (const float*)d_in[2];
  const float* W2  = (const float*)d_in[3];
  const float* pb2 = (const float*)d_in[4];
  float* out = (float*)d_out;
  long long Ntot = (long long)in_sizes[0] / (NKPT * 3);
  int blocks = (int)((Ntot + RPB - 1) / RPB);
  pose_encoder_wmma<<<blocks, RPB, 0, stream>>>(kp, W1, pb1, W2, pb2, out, Ntot);
}
